// DCRNN_25872882991477
// MI455X (gfx1250) — matrix operand order repaired and checked by reading, not compiled, not run on hardware
//
#include <hip/hip_runtime.h>

#define N_NODES 2048
#define HID 64
#define C2H 128
#define TOT_CH 640
#define BATCH 2
#define SEQ_LEN 3
#define NLAYER 2

typedef __bf16 bf16x16 __attribute__((ext_vector_type(16)));
typedef float f32x8 __attribute__((ext_vector_type(8)));
typedef unsigned int u32x4 __attribute__((ext_vector_type(4)));

union Frag {
  bf16x16 v;
  u32x4 q[2];
};

#define LDA_S 40  // padded LDS row stride (elements); 80B keeps b128 loads 16B-aligned
#define LDB_S 40

// C = A(MxK bf16, lda) * B(KxNc bf16, ldb); block tile 128x64, 8 waves,
// each wave owns a 32x32 sub-tile = 2x2 WMMA tiles (4 WMMAs per 32-deep K step,
// reusing 2 A-frags x 2 B-frags). gridDim.z batches with element strides aZ/bZ/cZ.
// Output: bf16 (Cb) or f32 (Cf), row stride ldc.
__global__ __launch_bounds__(256) void wmma_gemm_bf16(
    const __bf16* __restrict__ A, int lda, long aZ,
    const __bf16* __restrict__ B, int ldb, long bZ,
    __bf16* __restrict__ Cb, float* __restrict__ Cf, int ldc, long cZ,
    int K)
{
  __shared__ __bf16 As[128 * LDA_S];
  __shared__ __bf16 BsT[64 * LDB_S];  // transposed: BsT[n][k]

  const int tid  = threadIdx.x;
  const int lane = tid & 31;
  const int wid  = tid >> 5;
  const int wm   = wid & 3;    // wave rows: wm*32
  const int wn   = wid >> 2;   // wave cols: wn*32
  const int m0   = blockIdx.x * 128;
  const int n0   = blockIdx.y * 64;

  A += (size_t)blockIdx.z * aZ;
  B += (size_t)blockIdx.z * bZ;

  // cooperative A tile (128x32): row = tid>>1, col base = (tid&1)*16, 2 x b128
  const int la_r = tid >> 1;
  const int la_c = (tid & 1) << 4;
  // cooperative B tile (32x64): k = tid>>3, n = (tid&7)*8, 1 x b128 -> 8 b16 transposed
  const int lb_k = tid >> 3;
  const int lb_n = (tid & 7) << 3;

  // fragment coords
  const int fr  = lane & 15;           // row (A) / col (B) within 16-tile
  const int kbA = (lane >> 4) << 3;    // A frag: K groups {0-7,16-23} / {8-15,24-31}
  const int koB = (lane >> 4) << 3;    // B frag: K groups {0-7,16-23} / {8-15,24-31}

  f32x8 acc00 = {0.f, 0.f, 0.f, 0.f, 0.f, 0.f, 0.f, 0.f};
  f32x8 acc01 = acc00, acc10 = acc00, acc11 = acc00;

  for (int k0 = 0; k0 < K; k0 += 32) {
    const __bf16* ag = A + (size_t)(m0 + la_r) * lda + (k0 + la_c);
    u32x4 a0 = *(const u32x4*)(ag);
    u32x4 a1 = *(const u32x4*)(ag + 8);
    *(u32x4*)(&As[la_r * LDA_S + la_c])     = a0;
    *(u32x4*)(&As[la_r * LDA_S + la_c + 8]) = a1;

    u32x4 bvec = *(const u32x4*)(B + (size_t)(k0 + lb_k) * ldb + (n0 + lb_n));
    union { u32x4 u; __bf16 e[8]; } bb; bb.u = bvec;
#pragma unroll
    for (int i = 0; i < 8; ++i)
      BsT[(lb_n + i) * LDB_S + lb_k] = bb.e[i];

    if (k0 + 32 < K)  // next A tile -> global_prefetch
      __builtin_prefetch(ag + 32, 0, 3);

    __syncthreads();

    Frag fa0, fa1, fb0, fb1;
    {
      const int ar0 = (wm * 32 + fr) * LDA_S;
      fa0.q[0] = *(const u32x4*)(&As[ar0 + kbA]);
      fa0.q[1] = *(const u32x4*)(&As[ar0 + kbA + 16]);
      const int ar1 = (wm * 32 + 16 + fr) * LDA_S;
      fa1.q[0] = *(const u32x4*)(&As[ar1 + kbA]);
      fa1.q[1] = *(const u32x4*)(&As[ar1 + kbA + 16]);
      const int bc0 = (wn * 32 + fr) * LDB_S + koB;
      fb0.q[0] = *(const u32x4*)(&BsT[bc0]);
      fb0.q[1] = *(const u32x4*)(&BsT[bc0 + 16]);
      const int bc1 = (wn * 32 + 16 + fr) * LDB_S + koB;
      fb1.q[0] = *(const u32x4*)(&BsT[bc1]);
      fb1.q[1] = *(const u32x4*)(&BsT[bc1 + 16]);
    }

    acc00 = __builtin_amdgcn_wmma_f32_16x16x32_bf16(false, fa0.v, false, fb0.v, (short)0, acc00, false, false);
    acc01 = __builtin_amdgcn_wmma_f32_16x16x32_bf16(false, fa0.v, false, fb1.v, (short)0, acc01, false, false);
    acc10 = __builtin_amdgcn_wmma_f32_16x16x32_bf16(false, fa1.v, false, fb0.v, (short)0, acc10, false, false);
    acc11 = __builtin_amdgcn_wmma_f32_16x16x32_bf16(false, fa1.v, false, fb1.v, (short)0, acc11, false, false);

    __syncthreads();
  }

  // C/D layout: N = lane&15; VGPR r -> M = r (lanes 0-15) or r+8 (lanes 16-31)
  const int rbase = m0 + wm * 32 + ((lane >> 4) << 3);
  const int cbase = n0 + wn * 32 + fr;
  if (Cf) {
    Cf += (size_t)blockIdx.z * cZ;
    auto st = [&](f32x8 a, int mi, int ni) {
#pragma unroll
      for (int r = 0; r < 8; ++r)
        Cf[(size_t)(rbase + mi * 16 + r) * ldc + (cbase + ni * 16)] = a[r];
    };
    st(acc00, 0, 0); st(acc01, 0, 1); st(acc10, 1, 0); st(acc11, 1, 1);
  } else {
    Cb += (size_t)blockIdx.z * cZ;
    auto st = [&](f32x8 a, int mi, int ni) {
#pragma unroll
      for (int r = 0; r < 8; ++r)
        Cb[(size_t)(rbase + mi * 16 + r) * ldc + (cbase + ni * 16)] = (__bf16)a[r];
    };
    st(acc00, 0, 0); st(acc01, 0, 1); st(acc10, 1, 0); st(acc11, 1, 1);
  }
}

// Row-normalize adj (+1e-6 on diag, +1e-8 eps), emit bf16 A and A^T.
__global__ __launch_bounds__(256) void normalize_adj(
    const float* __restrict__ adj, __bf16* __restrict__ Abf,
    __bf16* __restrict__ ATbf)
{
  __shared__ float red[256];
  const int i = blockIdx.x;
  float s = 0.f;
  for (int j = threadIdx.x; j < N_NODES; j += 256) {
    float v = adj[(size_t)i * N_NODES + j];
    if (j == i) v += 1e-6f;
    s += v;
  }
  red[threadIdx.x] = s;
  __syncthreads();
  for (int off = 128; off > 0; off >>= 1) {
    if (threadIdx.x < off) red[threadIdx.x] += red[threadIdx.x + off];
    __syncthreads();
  }
  const float inv = 1.0f / (red[0] + 1e-8f);
  for (int j = threadIdx.x; j < N_NODES; j += 256) {
    float v = adj[(size_t)i * N_NODES + j];
    if (j == i) v += 1e-6f;
    __bf16 bv = (__bf16)(v * inv);
    Abf[(size_t)i * N_NODES + j] = bv;
    ATbf[(size_t)j * N_NODES + i] = bv;
  }
}

__global__ void zero_f32(float* p, int n) {
  int i = blockIdx.x * 256 + threadIdx.x;
  if (i < n) p[i] = 0.f;
}

__global__ void f32_to_bf16_k(const float* __restrict__ in, __bf16* __restrict__ out, int n) {
  int i = blockIdx.x * 256 + threadIdx.x;
  if (i < n) out[i] = (__bf16)in[i];
}

// xp[b,t,n,h] = inputs[b,t,n] * w[h] + bias[h]   (over B*SEQ*N*H threads)
__global__ void in_proj_k(const float* __restrict__ inp, const float* __restrict__ w,
                          const float* __restrict__ b, float* __restrict__ xp) {
  int i = blockIdx.x * 256 + threadIdx.x;
  int hh = i & (HID - 1);
  int node = i >> 6;
  xp[i] = inp[node] * w[hh] + b[hh];
}

// Z[b,n,0:128] = bf16(concat(x, h)); x may be null (zeros). i over B*N*128.
__global__ void build_xh_k(const float* __restrict__ x, unsigned long long xbs,
                           const float* __restrict__ h, __bf16* __restrict__ Z) {
  int i = blockIdx.x * 256 + threadIdx.x;
  int c = i & (C2H - 1);
  int n = (i >> 7) & (N_NODES - 1);
  int b = i >> 18;
  float v;
  if (c < HID)
    v = x ? x[(size_t)b * xbs + (size_t)n * HID + c] : 0.f;
  else
    v = h[((size_t)b * N_NODES + n) * HID + (c - HID)];
  Z[((size_t)b * N_NODES + n) * TOT_CH + c] = (__bf16)v;
}

// r,u = sigmoid(G + gb); store u; Z[b,n,64+c] = bf16(r*h).  i over B*N*64.
__global__ void gates_k(const float* __restrict__ G, const float* __restrict__ gb,
                        const float* __restrict__ h, float* __restrict__ U,
                        __bf16* __restrict__ Z) {
  int i = blockIdx.x * 256 + threadIdx.x;
  int c = i & (HID - 1);
  int n = (i >> 6) & (N_NODES - 1);
  int b = i >> 17;
  size_t gbase = ((size_t)b * N_NODES + n) * C2H;
  float r = 1.f / (1.f + __expf(-(G[gbase + c] + gb[c])));
  float u = 1.f / (1.f + __expf(-(G[gbase + HID + c] + gb[HID + c])));
  size_t hb = ((size_t)b * N_NODES + n) * HID + c;
  U[hb] = u;
  Z[((size_t)b * N_NODES + n) * TOT_CH + HID + c] = (__bf16)(r * h[hb]);
}

// h = u*h + (1-u)*tanh(C + cb).  i over B*N*64; C/U/h share (b,n,c) layout.
__global__ void cand_k(const float* __restrict__ C, const float* __restrict__ cb,
                       const float* __restrict__ U, float* __restrict__ h) {
  int i = blockIdx.x * 256 + threadIdx.x;
  int c = i & (HID - 1);
  float cv = tanhf(C[i] + cb[c]);
  float u = U[i];
  h[i] = u * h[i] + (1.f - u) * cv;
}

// out[b*N+n] = h1[b,n,:] . w + b0
__global__ void out_proj_k(const float* __restrict__ h1, const float* __restrict__ w,
                           const float* __restrict__ b, float* __restrict__ out) {
  int i = blockIdx.x * 256 + threadIdx.x;
  if (i >= BATCH * N_NODES) return;
  float s = b[0];
  const float* hp = h1 + (size_t)i * HID;
#pragma unroll 8
  for (int k = 0; k < HID; ++k) s += hp[k] * w[k];
  out[i] = s;
}

extern "C" void kernel_launch(void* const* d_in, const int* in_sizes, int n_in,
                              void* d_out, int out_size, void* d_ws, size_t ws_size,
                              hipStream_t stream) {
  const float* inputs = (const float*)d_in[0];
  const float* adj    = (const float*)d_in[1];
  const float* in_w   = (const float*)d_in[2];
  const float* in_b   = (const float*)d_in[3];
  const float* eg_w   = (const float*)d_in[4];
  const float* eg_b   = (const float*)d_in[5];
  const float* ec_w   = (const float*)d_in[6];
  const float* ec_b   = (const float*)d_in[7];
  const float* dg_w   = (const float*)d_in[8];
  const float* dg_b   = (const float*)d_in[9];
  const float* dc_w   = (const float*)d_in[10];
  const float* dc_b   = (const float*)d_in[11];
  const float* ow     = (const float*)d_in[12];
  const float* ob     = (const float*)d_in[13];
  float* out = (float*)d_out;

  char* p = (char*)d_ws;
  auto carve = [&](size_t bytes) {
    char* r = p;
    p += (bytes + 255) & ~(size_t)255;
    return (void*)r;
  };
  __bf16* Abf  = (__bf16*)carve((size_t)N_NODES * N_NODES * 2);
  __bf16* ATbf = (__bf16*)carve((size_t)N_NODES * N_NODES * 2);
  __bf16* Wge  = (__bf16*)carve((size_t)NLAYER * TOT_CH * C2H * 2);
  __bf16* Wce  = (__bf16*)carve((size_t)NLAYER * TOT_CH * HID * 2);
  __bf16* Wgd  = (__bf16*)carve((size_t)NLAYER * TOT_CH * C2H * 2);
  __bf16* Wcd  = (__bf16*)carve((size_t)NLAYER * TOT_CH * HID * 2);
  __bf16* Z    = (__bf16*)carve((size_t)BATCH * N_NODES * TOT_CH * 2);
  float*  Gbuf = (float*)carve((size_t)BATCH * N_NODES * C2H * 4);
  float*  Cbuf = (float*)carve((size_t)BATCH * N_NODES * HID * 4);
  float*  Ubuf = (float*)carve((size_t)BATCH * N_NODES * HID * 4);
  float*  Hbuf = (float*)carve((size_t)NLAYER * BATCH * N_NODES * HID * 4);
  float*  XP   = (float*)carve((size_t)BATCH * SEQ_LEN * N_NODES * HID * 4);

  // ---- prep ----
  normalize_adj<<<N_NODES, 256, 0, stream>>>(adj, Abf, ATbf);
  {
    int ng = NLAYER * TOT_CH * C2H, nc = NLAYER * TOT_CH * HID;
    f32_to_bf16_k<<<(ng + 255) / 256, 256, 0, stream>>>(eg_w, Wge, ng);
    f32_to_bf16_k<<<(nc + 255) / 256, 256, 0, stream>>>(ec_w, Wce, nc);
    f32_to_bf16_k<<<(ng + 255) / 256, 256, 0, stream>>>(dg_w, Wgd, ng);
    f32_to_bf16_k<<<(nc + 255) / 256, 256, 0, stream>>>(dc_w, Wcd, nc);
  }
  {
    int nh = NLAYER * BATCH * N_NODES * HID;
    zero_f32<<<(nh + 255) / 256, 256, 0, stream>>>(Hbuf, nh);
    int nx = BATCH * SEQ_LEN * N_NODES * HID;
    in_proj_k<<<nx / 256, 256, 0, stream>>>(inputs, in_w, in_b, XP);
  }

  auto gemm = [&](const __bf16* A, int lda, long aZ,
                  const __bf16* B, int ldb, long bZ,
                  __bf16* Cb_, float* Cf_, int ldc, long cZ,
                  int Nc, int K, int nz) {
    dim3 g(N_NODES / 128, Nc / 64, nz);
    wmma_gemm_bf16<<<g, 256, 0, stream>>>(A, lda, aZ, B, ldb, bZ, Cb_, Cf_, ldc, cZ, K);
  };

  const long NT = (long)N_NODES * TOT_CH;

  // cols of Z per batch: [x|h*: 0..127][A@: 128][A^2@: 256][AT@: 384][AT^2@: 512]
  auto diffusion = [&]() {
    gemm(Abf,  N_NODES, 0, Z,       TOT_CH, NT, Z + 128, nullptr, TOT_CH, NT, C2H, N_NODES, BATCH);
    gemm(Abf,  N_NODES, 0, Z + 128, TOT_CH, NT, Z + 256, nullptr, TOT_CH, NT, C2H, N_NODES, BATCH);
    gemm(ATbf, N_NODES, 0, Z,       TOT_CH, NT, Z + 384, nullptr, TOT_CH, NT, C2H, N_NODES, BATCH);
    gemm(ATbf, N_NODES, 0, Z + 384, TOT_CH, NT, Z + 512, nullptr, TOT_CH, NT, C2H, N_NODES, BATCH);
  };

  auto cell = [&](const float* xptr, unsigned long long xbs, float* h,
                  const __bf16* Wg, const float* gb,
                  const __bf16* Wc, const float* cb) {
    build_xh_k<<<(BATCH * N_NODES * C2H) / 256, 256, 0, stream>>>(xptr, xbs, h, Z);
    diffusion();
    gemm(Z, TOT_CH, NT, Wg, C2H, 0,
         nullptr, Gbuf, C2H, (long)N_NODES * C2H, C2H, TOT_CH, BATCH);
    gates_k<<<(BATCH * N_NODES * HID) / 256, 256, 0, stream>>>(Gbuf, gb, h, Ubuf, Z);
    diffusion();
    gemm(Z, TOT_CH, NT, Wc, HID, 0,
         nullptr, Cbuf, HID, (long)N_NODES * HID, HID, TOT_CH, BATCH);
    cand_k<<<(BATCH * N_NODES * HID) / 256, 256, 0, stream>>>(Cbuf, cb, Ubuf, h);
  };

  const size_t hstride = (size_t)BATCH * N_NODES * HID;

  // ---- encoder ----
  for (int t = 0; t < SEQ_LEN; ++t) {
    for (int l = 0; l < NLAYER; ++l) {
      const float* x;
      unsigned long long xbs;
      if (l == 0) { x = XP + (size_t)t * N_NODES * HID; xbs = (unsigned long long)SEQ_LEN * N_NODES * HID; }
      else        { x = Hbuf + (size_t)(l - 1) * hstride; xbs = (unsigned long long)N_NODES * HID; }
      cell(x, xbs, Hbuf + (size_t)l * hstride,
           Wge + (size_t)l * TOT_CH * C2H, eg_b + (size_t)l * C2H,
           Wce + (size_t)l * TOT_CH * HID, ec_b + (size_t)l * HID);
    }
  }

  // ---- decoder (PRED == 1) ----
  for (int l = 0; l < NLAYER; ++l) {
    const float* x = (l == 0) ? nullptr : Hbuf + (size_t)(l - 1) * hstride;
    unsigned long long xbs = (unsigned long long)N_NODES * HID;
    cell(x, xbs, Hbuf + (size_t)l * hstride,
         Wgd + (size_t)l * TOT_CH * C2H, dg_b + (size_t)l * C2H,
         Wcd + (size_t)l * TOT_CH * HID, dc_b + (size_t)l * HID);
  }
  out_proj_k<<<(BATCH * N_NODES + 255) / 256, 256, 0, stream>>>(
      Hbuf + hstride, ow, ob, out);
}
